// DecoderLayer_51685636440598
// MI455X (gfx1250) — matrix operand order repaired and checked by reading, not compiled, not run on hardware
//
#include <hip/hip_runtime.h>

// ---------------------------------------------------------------------------
// CDNA5 (gfx1250) decoder layer. All matmuls via v_wmma_f32_16x16x32_bf16.
// Compute-bound (~620 GFLOP vs ~0.3GB HBM traffic at 23.3 TB/s) -> bf16 WMMA
// with fp32 accumulation. This revision adds amdgpu_waves_per_eu(1) on the
// WMMA kernels (relaxes the VGPR-pressure target that was collapsing all
// fragment buffers onto one register octet and serializing ds_load->wait->wmma)
// and widens the flash S-burst to 8 independent B-fragments per wait.
// ---------------------------------------------------------------------------

typedef __bf16 bf16_t;
typedef __attribute__((ext_vector_type(16))) __bf16 v16bf;
typedef __attribute__((ext_vector_type(8)))  __bf16 v8bf;
typedef __attribute__((ext_vector_type(8)))  float  v8f;

#define WMMA_BF16(a, b, c) \
    __builtin_amdgcn_wmma_f32_16x16x32_bf16(false, (a), false, (b), (short)0, (c), false, false)

__device__ __forceinline__ v16bf ld_bf16x16(const bf16_t* p) {
    // 16 contiguous halfs as two 16B loads (16B-aligned callers only)
    v8bf lo = *(const v8bf*)p;
    v8bf hi = *(const v8bf*)(p + 8);
    return __builtin_shufflevector(lo, hi, 0,1,2,3,4,5,6,7,8,9,10,11,12,13,14,15);
}
__device__ __forceinline__ v16bf ld_bf16x16_gap(const bf16_t* p) {
    // A-fragment pattern: 8 halfs at p, 8 halfs at p+16
    v8bf lo = *(const v8bf*)p;
    v8bf hi = *(const v8bf*)(p + 16);
    return __builtin_shufflevector(lo, hi, 0,1,2,3,4,5,6,7,8,9,10,11,12,13,14,15);
}

// ---------------------------------------------------------------------------
// fp32 -> bf16 weight conversion with optional column concat (QKV packing)
// ---------------------------------------------------------------------------
__global__ void cvt_f32_bf16_strided(const float* __restrict__ src, bf16_t* __restrict__ dst,
                                     int rows, int cols, int dstStride, int colOff) {
    long long idx = (long long)blockIdx.x * blockDim.x + threadIdx.x;
    long long total = (long long)rows * cols;
    if (idx >= total) return;
    int r = (int)(idx / cols), c = (int)(idx % cols);
    dst[(size_t)r * dstStride + colOff + c] = (bf16_t)src[idx];
}

// ---------------------------------------------------------------------------
// RMSNorm: one row (D=2048) per 256-thread block, bf16 output
// ---------------------------------------------------------------------------
__global__ __launch_bounds__(256)
void rmsnorm_bf16(const float* __restrict__ x, const float* __restrict__ g,
                  bf16_t* __restrict__ out, int D_) {
    __shared__ float red[8];
    const int row = blockIdx.x, tid = threadIdx.x;
    const float* xr = x + (size_t)row * D_;
    float ss = 0.f;
    const float4* x4 = (const float4*)xr;
    for (int i = tid; i < D_ / 4; i += 256) {
        float4 v = x4[i];
        ss += v.x * v.x + v.y * v.y + v.z * v.z + v.w * v.w;
    }
    for (int m = 16; m; m >>= 1) ss += __shfl_xor(ss, m);
    if ((tid & 31) == 0) red[tid >> 5] = ss;
    __syncthreads();
    float tot = 0.f;
    #pragma unroll
    for (int i = 0; i < 8; ++i) tot += red[i];
    float sc = rsqrtf(tot / (float)D_ + 1e-6f);
    for (int i = tid; i < D_; i += 256)
        out[(size_t)row * D_ + i] = (bf16_t)(xr[i] * sc * g[i]);
}

// ---------------------------------------------------------------------------
// Tiled bf16 WMMA GEMM: C[M,N] = A[M,K] @ W[K,N]
//   block = 256 threads (8 waves), tile 256x128, K-step 32
//   wave grid 4x2 -> each wave: 64x64 = 4x4 WMMA fragments (each frag reused 4x)
//   global->LDS staging is register double-buffered across K-steps.
// EPI: 0 = store f32
//      2 = store f32 + residual (aux)
//      3 = store bf16 of silu(aux) * acc   (fused SwiGLU)
// ---------------------------------------------------------------------------
template <int EPI>
__global__ __launch_bounds__(256)
__attribute__((amdgpu_waves_per_eu(1)))
void gemm_bf16_wmma(const bf16_t* __restrict__ A, const bf16_t* __restrict__ W,
                    int M, int N, int K,
                    float* __restrict__ outF, bf16_t* __restrict__ outH,
                    const float* __restrict__ aux) {
    __shared__ __align__(16) bf16_t As[256][40];   // [m][k], 80B rows (16B aligned)
    __shared__ __align__(16) bf16_t Bs[128][40];   // [n][k] (transposed)

    const int tid  = threadIdx.x;
    const int lane = tid & 31, wv = tid >> 5;
    const int lh   = lane & 15, hs = lane >> 4;
    const int mw   = wv >> 1,   nw = wv & 1;
    const int m0   = blockIdx.y * 256, n0 = blockIdx.x * 128;

    v8f acc[4][4];
    #pragma unroll
    for (int i = 0; i < 4; ++i)
        #pragma unroll
        for (int j = 0; j < 4; ++j) acc[i][j] = (v8f)0.f;

    // staging assignments: A row per thread (256 rows x 32 halfs),
    // B: thread -> (k = tid>>3, 16 cols starting at (tid&7)*16)
    const int lb_k = tid >> 3;
    const int lb_n = (tid & 7) * 16;
    const bf16_t* arow = A + (size_t)(m0 + tid) * K;

    uint4 areg[4], breg[2];
    #pragma unroll
    for (int i = 0; i < 4; ++i) areg[i] = *(const uint4*)(arow + i * 8);
    {
        const bf16_t* bp = W + (size_t)lb_k * N + n0 + lb_n;
        breg[0] = *(const uint4*)bp;
        breg[1] = *(const uint4*)(bp + 8);
    }

    for (int kt = 0; kt < K; kt += 32) {
        __syncthreads();   // previous iteration finished reading LDS
        #pragma unroll
        for (int i = 0; i < 4; ++i) *(uint4*)&As[tid][i * 8] = areg[i];
        {
            const bf16_t* h0 = (const bf16_t*)&breg[0];
            const bf16_t* h1 = (const bf16_t*)&breg[1];
            #pragma unroll
            for (int i = 0; i < 8; ++i) {
                Bs[lb_n + i][lb_k]     = h0[i];
                Bs[lb_n + 8 + i][lb_k] = h1[i];
            }
        }
        __syncthreads();

        // prefetch next K-step into registers (overlaps WMMA compute below)
        if (kt + 32 < K) {
            const bf16_t* ap2 = arow + kt + 32;
            #pragma unroll
            for (int i = 0; i < 4; ++i) areg[i] = *(const uint4*)(ap2 + i * 8);
            const bf16_t* bp2 = W + (size_t)(kt + 32 + lb_k) * N + n0 + lb_n;
            breg[0] = *(const uint4*)bp2;
            breg[1] = *(const uint4*)(bp2 + 8);
            if (kt + 64 < K) {   // global_prefetch_b8 one tile further ahead
                __builtin_prefetch(arow + kt + 64, 0, 1);
                __builtin_prefetch(W + (size_t)(kt + 64 + lb_k) * N + n0 + lb_n, 0, 1);
            }
        }

        // all 8 fragments in distinct registers -> one wait, 16 back-to-back WMMAs
        v16bf afr0 = ld_bf16x16(&As[mw * 64 + 0  + lh][hs * 16]);
        v16bf afr1 = ld_bf16x16(&As[mw * 64 + 16 + lh][hs * 16]);
        v16bf afr2 = ld_bf16x16(&As[mw * 64 + 32 + lh][hs * 16]);
        v16bf afr3 = ld_bf16x16(&As[mw * 64 + 48 + lh][hs * 16]);
        v16bf bfr0 = ld_bf16x16(&Bs[nw * 64 + 0  + lh][hs * 16]);
        v16bf bfr1 = ld_bf16x16(&Bs[nw * 64 + 16 + lh][hs * 16]);
        v16bf bfr2 = ld_bf16x16(&Bs[nw * 64 + 32 + lh][hs * 16]);
        v16bf bfr3 = ld_bf16x16(&Bs[nw * 64 + 48 + lh][hs * 16]);

        acc[0][0] = WMMA_BF16(afr0, bfr0, acc[0][0]);
        acc[0][1] = WMMA_BF16(afr0, bfr1, acc[0][1]);
        acc[0][2] = WMMA_BF16(afr0, bfr2, acc[0][2]);
        acc[0][3] = WMMA_BF16(afr0, bfr3, acc[0][3]);
        acc[1][0] = WMMA_BF16(afr1, bfr0, acc[1][0]);
        acc[1][1] = WMMA_BF16(afr1, bfr1, acc[1][1]);
        acc[1][2] = WMMA_BF16(afr1, bfr2, acc[1][2]);
        acc[1][3] = WMMA_BF16(afr1, bfr3, acc[1][3]);
        acc[2][0] = WMMA_BF16(afr2, bfr0, acc[2][0]);
        acc[2][1] = WMMA_BF16(afr2, bfr1, acc[2][1]);
        acc[2][2] = WMMA_BF16(afr2, bfr2, acc[2][2]);
        acc[2][3] = WMMA_BF16(afr2, bfr3, acc[2][3]);
        acc[3][0] = WMMA_BF16(afr3, bfr0, acc[3][0]);
        acc[3][1] = WMMA_BF16(afr3, bfr1, acc[3][1]);
        acc[3][2] = WMMA_BF16(afr3, bfr2, acc[3][2]);
        acc[3][3] = WMMA_BF16(afr3, bfr3, acc[3][3]);
    }

    // --- epilogue (C layout: M = r + 8*hs, N = lh) ---
    #pragma unroll
    for (int ms = 0; ms < 4; ++ms)
        #pragma unroll
        for (int ns = 0; ns < 4; ++ns)
            #pragma unroll
            for (int r = 0; r < 8; ++r) {
                int row = m0 + mw * 64 + ms * 16 + r + 8 * hs;
                int col = n0 + nw * 64 + ns * 16 + lh;
                size_t idx = (size_t)row * N + col;
                float v = acc[ms][ns][r];
                if (EPI == 0) {
                    outF[idx] = v;
                } else if (EPI == 2) {
                    outF[idx] = v + aux[idx];
                } else { // EPI == 3: silu(gate) * up -> bf16
                    float gt = aux[idx];
                    outH[idx] = (bf16_t)((gt / (1.f + __expf(-gt))) * v);
                }
            }
}

// ---------------------------------------------------------------------------
// RoPE + split: qkv fp32 (B*T rows of 6144 = [q|k|v]) -> q/k/v bf16 (b,h,t,hd)
// position_ids is int64 (reference: arange(B*T))
// ---------------------------------------------------------------------------
__global__ void rope_split(const float* __restrict__ qkv, const long long* __restrict__ pos_ids,
                           bf16_t* __restrict__ qh, bf16_t* __restrict__ kh,
                           bf16_t* __restrict__ vh, int B_, int T_, int Hn) {
    long long idx = (long long)blockIdx.x * blockDim.x + threadIdx.x;
    long long total = (long long)B_ * T_ * Hn * 64;
    if (idx >= total) return;
    int j = (int)(idx & 63);
    long long rest = idx >> 6;
    int h = (int)(rest % Hn);
    long long bt = rest / Hn;          // b*T + t
    long long pos = pos_ids[bt];
    float inv = __powf(10000.0f, -(float)(2 * j) * (1.0f / 128.0f));
    float ang = (float)pos * inv;
    float s, c;
    __sincosf(ang, &s, &c);
    const float* rp = qkv + (size_t)bt * 6144 + h * 128;
    float q1 = rp[j],        q2 = rp[j + 64];
    float k1 = rp[2048 + j], k2 = rp[2048 + j + 64];
    float v1 = rp[4096 + j], v2 = rp[4096 + j + 64];
    int b = (int)(bt / T_), t = (int)(bt % T_);
    size_t orow = (((size_t)b * Hn + h) * T_ + t) * 128;
    qh[orow + j]      = (bf16_t)(q1 * c - q2 * s);
    qh[orow + j + 64] = (bf16_t)(q2 * c + q1 * s);
    kh[orow + j]      = (bf16_t)(k1 * c - k2 * s);
    kh[orow + j + 64] = (bf16_t)(k2 * c + k1 * s);
    vh[orow + j]      = (bf16_t)v1;
    vh[orow + j + 64] = (bf16_t)v2;
}

// ---------------------------------------------------------------------------
// Causal flash attention, bf16 WMMA, fp32 online softmax.
// grid = (T/64, B*H), block = 128 threads (4 waves, 16 q-rows each)
// ---------------------------------------------------------------------------
__global__ __launch_bounds__(128)
__attribute__((amdgpu_waves_per_eu(1)))
void flash_attn_wmma(const bf16_t* __restrict__ Qm, const bf16_t* __restrict__ Km,
                     const bf16_t* __restrict__ Vm, bf16_t* __restrict__ Oout,
                     int T_, int Hn) {
    __shared__ __align__(16) bf16_t Ks[64][136];     // [key][hd]
    __shared__ __align__(16) bf16_t Vt[128][72];     // [hd][key] (transposed)
    __shared__ __align__(16) bf16_t Ps[4][16][72];   // per-wave P strip [row][key]

    const int tid = threadIdx.x, lane = tid & 31, wv = tid >> 5;
    const int lh = lane & 15, hs = lane >> 4;
    const int bh = blockIdx.y;
    const int b = bh / Hn, h = bh % Hn;
    const int q0 = blockIdx.x * 64;
    const size_t base = (size_t)bh * T_ * 128;
    const bf16_t* Qb = Qm + base;
    const bf16_t* Kb = Km + base;
    const bf16_t* Vb = Vm + base;

    // Q fragments for this wave's 16-row strip (A layout)
    const int qr = q0 + wv * 16 + lh;
    v16bf qa[4];
    {
        const bf16_t* qp = Qb + (size_t)qr * 128;
        #pragma unroll
        for (int kc = 0; kc < 4; ++kc)
            qa[kc] = ld_bf16x16(qp + kc * 32 + hs * 16);
    }

    v8f oacc[8];
    #pragma unroll
    for (int i = 0; i < 8; ++i) oacc[i] = (v8f)0.f;
    float mrow[8], lrow[8];
    #pragma unroll
    for (int r = 0; r < 8; ++r) { mrow[r] = -3.0e38f; lrow[r] = 0.f; }

    const float scale = 0.08838834764831845f;   // 1/sqrt(128)

    for (int kb = 0; kb <= q0; kb += 64) {
        // stage K (row-major) and V (transposed) tiles
        {
            int r = tid >> 1, cb = (tid & 1) * 64;
            const bf16_t* kp = Kb + (size_t)(kb + r) * 128 + cb;
            #pragma unroll
            for (int i = 0; i < 8; ++i)
                *(uint4*)&Ks[r][cb + i * 8] = *(const uint4*)(kp + i * 8);
            const bf16_t* vp = Vb + (size_t)(kb + r) * 128 + cb;
            #pragma unroll
            for (int i = 0; i < 8; ++i) {
                uint4 tv = *(const uint4*)(vp + i * 8);
                const bf16_t* th = (const bf16_t*)&tv;
                #pragma unroll
                for (int j = 0; j < 8; ++j) Vt[cb + i * 8 + j][r] = th[j];
            }
        }
        __syncthreads();

        // S = Q * K^T: 8 independent B-fragments in flight per burst
        // (two K-chunks x four key subtiles), then 8 WMMAs.
        v8f sfr[4];
        #pragma unroll
        for (int n = 0; n < 4; ++n) sfr[n] = (v8f)0.f;
        #pragma unroll
        for (int kp2 = 0; kp2 < 2; ++kp2) {
            const int kcA = kp2 * 2, kcB = kp2 * 2 + 1;
            v16bf a0 = ld_bf16x16(&Ks[ 0 + lh][kcA * 32 + hs * 16]);
            v16bf a1 = ld_bf16x16(&Ks[16 + lh][kcA * 32 + hs * 16]);
            v16bf a2 = ld_bf16x16(&Ks[32 + lh][kcA * 32 + hs * 16]);
            v16bf a3 = ld_bf16x16(&Ks[48 + lh][kcA * 32 + hs * 16]);
            v16bf b0 = ld_bf16x16(&Ks[ 0 + lh][kcB * 32 + hs * 16]);
            v16bf b1 = ld_bf16x16(&Ks[16 + lh][kcB * 32 + hs * 16]);
            v16bf b2 = ld_bf16x16(&Ks[32 + lh][kcB * 32 + hs * 16]);
            v16bf b3 = ld_bf16x16(&Ks[48 + lh][kcB * 32 + hs * 16]);
            sfr[0] = WMMA_BF16(qa[kcA], a0, sfr[0]);
            sfr[1] = WMMA_BF16(qa[kcA], a1, sfr[1]);
            sfr[2] = WMMA_BF16(qa[kcA], a2, sfr[2]);
            sfr[3] = WMMA_BF16(qa[kcA], a3, sfr[3]);
            sfr[0] = WMMA_BF16(qa[kcB], b0, sfr[0]);
            sfr[1] = WMMA_BF16(qa[kcB], b1, sfr[1]);
            sfr[2] = WMMA_BF16(qa[kcB], b2, sfr[2]);
            sfr[3] = WMMA_BF16(qa[kcB], b3, sfr[3]);
        }

        // online softmax per owned row (C layout row = r + 8*hs)
        #pragma unroll
        for (int r = 0; r < 8; ++r) {
            const int rq = q0 + wv * 16 + r + 8 * hs;
            float sv[4], mloc = -3.0e38f;
            #pragma unroll
            for (int n = 0; n < 4; ++n) {
                float v = sfr[n][r] * scale;
                int col = kb + n * 16 + lh;
                if (col > rq) v = -10000.0f;   // causal mask (matches reference)
                sv[n] = v;
                mloc = fmaxf(mloc, v);
            }
            for (int m = 8; m; m >>= 1) mloc = fmaxf(mloc, __shfl_xor(mloc, m));
            float mn = fmaxf(mrow[r], mloc);
            float alpha = __expf(mrow[r] - mn);
            mrow[r] = mn;
            float rs = 0.f;
            #pragma unroll
            for (int n = 0; n < 4; ++n) {
                float p = __expf(sv[n] - mn);
                rs += p;
                Ps[wv][r + 8 * hs][n * 16 + lh] = (bf16_t)p;
            }
            for (int m = 8; m; m >>= 1) rs += __shfl_xor(rs, m);
            lrow[r] = lrow[r] * alpha + rs;
            #pragma unroll
            for (int hd = 0; hd < 8; ++hd) oacc[hd][r] *= alpha;
        }
        __syncthreads();   // P visible + LDS ordering before P*V

        // O += P * V: V fragments batched 4 at a time into distinct regs
        #pragma unroll
        for (int kc2 = 0; kc2 < 2; ++kc2) {
            v16bf pa = ld_bf16x16(&Ps[wv][lh][kc2 * 32 + hs * 16]);
            #pragma unroll
            for (int g = 0; g < 2; ++g) {
                v16bf v0 = ld_bf16x16(&Vt[(g * 4 + 0) * 16 + lh][kc2 * 32 + hs * 16]);
                v16bf v1 = ld_bf16x16(&Vt[(g * 4 + 1) * 16 + lh][kc2 * 32 + hs * 16]);
                v16bf v2 = ld_bf16x16(&Vt[(g * 4 + 2) * 16 + lh][kc2 * 32 + hs * 16]);
                v16bf v3 = ld_bf16x16(&Vt[(g * 4 + 3) * 16 + lh][kc2 * 32 + hs * 16]);
                oacc[g * 4 + 0] = WMMA_BF16(pa, v0, oacc[g * 4 + 0]);
                oacc[g * 4 + 1] = WMMA_BF16(pa, v1, oacc[g * 4 + 1]);
                oacc[g * 4 + 2] = WMMA_BF16(pa, v2, oacc[g * 4 + 2]);
                oacc[g * 4 + 3] = WMMA_BF16(pa, v3, oacc[g * 4 + 3]);
            }
        }
        __syncthreads();   // before next tile overwrites Ks/Vt
    }

    // normalize and write (b, t, h*128+hd) bf16 for the Wo GEMM
    #pragma unroll
    for (int r = 0; r < 8; ++r) {
        float inv = 1.0f / lrow[r];
        int t = q0 + wv * 16 + r + 8 * hs;
        size_t orow = ((size_t)b * T_ + t) * (size_t)(Hn * 128) + h * 128;
        #pragma unroll
        for (int hd = 0; hd < 8; ++hd)
            Oout[orow + hd * 16 + lh] = (bf16_t)(oacc[hd][r] * inv);
    }
}

// ---------------------------------------------------------------------------
// Host orchestration
// ---------------------------------------------------------------------------
extern "C" void kernel_launch(void* const* d_in, const int* in_sizes, int n_in,
                              void* d_out, int out_size, void* d_ws, size_t ws_size,
                              hipStream_t stream) {
    (void)in_sizes; (void)n_in; (void)out_size;
    const int B = 2, T = 2048, D = 2048, H = 16, FF = 8192;
    const int M = B * T;   // 4096 tokens

    const float*     x    = (const float*)d_in[0];
    const long long* pids = (const long long*)d_in[1];
    const float*     Wq   = (const float*)d_in[2];
    const float*     Wk   = (const float*)d_in[3];
    const float*     Wv   = (const float*)d_in[4];
    const float*     Wo   = (const float*)d_in[5];
    const float*     Wg   = (const float*)d_in[6];
    const float*     Wu   = (const float*)d_in[7];
    const float*     Wd   = (const float*)d_in[8];
    const float*     g1   = (const float*)d_in[9];
    const float*     g2   = (const float*)d_in[10];
    float* out = (float*)d_out;
    char*  ws  = (char*)d_ws;

    // ---- workspace layout (total 384 MB; region A phase-overlapped) ----
    size_t off = 0;
    auto take = [&](size_t bytes) { size_t r = off; off += (bytes + 255) & ~(size_t)255; return r; };
    const size_t oWqkv = take((size_t)D * 3 * D * 2);       // 2048x6144 bf16
    const size_t oWo   = take((size_t)D * D * 2);
    const size_t oWg   = take((size_t)D * FF * 2);
    const size_t oWu   = take((size_t)D * FF * 2);
    const size_t oWd   = take((size_t)FF * D * 2);
    const size_t oXn   = take((size_t)M * D * 2);
    const size_t oX1   = take((size_t)M * D * 4);
    const size_t oHb   = take((size_t)M * D * 2);
    const size_t oRegA = take(201326592);                   // max(phase1, phase2)
    (void)ws_size;
    // phase 1 (attention) inside region A
    const size_t oQKV  = oRegA;                              // M x 6144 f32 (100.7MB)
    const size_t oQh   = oQKV  + (size_t)M * 3 * D * 4;      // M x D bf16
    const size_t oKh   = oQh   + (size_t)M * D * 2;
    const size_t oVh   = oKh   + (size_t)M * D * 2;
    const size_t oAttn = oVh   + (size_t)M * D * 2;
    // phase 2 (MLP) inside region A — attention scratch is dead by then
    const size_t oGate = oRegA;                              // M x FF f32 (134.2MB)
    const size_t oAct  = oGate + (size_t)M * FF * 4;         // M x FF bf16

    bf16_t* Wqkv_h = (bf16_t*)(ws + oWqkv);
    bf16_t* Wo_h   = (bf16_t*)(ws + oWo);
    bf16_t* Wg_h   = (bf16_t*)(ws + oWg);
    bf16_t* Wu_h   = (bf16_t*)(ws + oWu);
    bf16_t* Wd_h   = (bf16_t*)(ws + oWd);
    bf16_t* xn     = (bf16_t*)(ws + oXn);
    float*  x1     = (float*)(ws + oX1);
    bf16_t* hb     = (bf16_t*)(ws + oHb);
    float*  qkvF   = (float*)(ws + oQKV);
    bf16_t* qh     = (bf16_t*)(ws + oQh);
    bf16_t* kh     = (bf16_t*)(ws + oKh);
    bf16_t* vh     = (bf16_t*)(ws + oVh);
    bf16_t* attn   = (bf16_t*)(ws + oAttn);
    float*  gateF  = (float*)(ws + oGate);
    bf16_t* actB   = (bf16_t*)(ws + oAct);

    auto cdiv = [](long long a, long long b) { return (unsigned)((a + b - 1) / b); };

    // 1) weights -> bf16 (QKV packed into one 2048x6144 matrix)
    cvt_f32_bf16_strided<<<cdiv((long long)D * D, 256), 256, 0, stream>>>(Wq, Wqkv_h, D, D, 3 * D, 0);
    cvt_f32_bf16_strided<<<cdiv((long long)D * D, 256), 256, 0, stream>>>(Wk, Wqkv_h, D, D, 3 * D, D);
    cvt_f32_bf16_strided<<<cdiv((long long)D * D, 256), 256, 0, stream>>>(Wv, Wqkv_h, D, D, 3 * D, 2 * D);
    cvt_f32_bf16_strided<<<cdiv((long long)D * D, 256), 256, 0, stream>>>(Wo, Wo_h, D, D, D, 0);
    cvt_f32_bf16_strided<<<cdiv((long long)D * FF, 256), 256, 0, stream>>>(Wg, Wg_h, D, FF, FF, 0);
    cvt_f32_bf16_strided<<<cdiv((long long)D * FF, 256), 256, 0, stream>>>(Wu, Wu_h, D, FF, FF, 0);
    cvt_f32_bf16_strided<<<cdiv((long long)FF * D, 256), 256, 0, stream>>>(Wd, Wd_h, FF, D, D, 0);

    // 2) xn = rmsnorm(x, g1)
    rmsnorm_bf16<<<M, 256, 0, stream>>>(x, g1, xn, D);

    // 3) qkv = xn @ [Wq|Wk|Wv]
    gemm_bf16_wmma<0><<<dim3(3 * D / 128, M / 256), 256, 0, stream>>>(
        xn, Wqkv_h, M, 3 * D, D, qkvF, nullptr, nullptr);

    // 4) RoPE + split into (b,h,t,hd) bf16
    rope_split<<<cdiv((long long)B * T * H * 64, 256), 256, 0, stream>>>(
        qkvF, pids, qh, kh, vh, B, T, H);

    // 5) causal flash attention
    flash_attn_wmma<<<dim3(T / 64, B * H), 128, 0, stream>>>(qh, kh, vh, attn, T, H);

    // 6) x1 = attn @ Wo + x
    gemm_bf16_wmma<2><<<dim3(D / 128, M / 256), 256, 0, stream>>>(
        attn, Wo_h, M, D, D, x1, nullptr, x);

    // 7) h = rmsnorm(x1, g2)
    rmsnorm_bf16<<<M, 256, 0, stream>>>(x1, g2, hb, D);

    // 8) gate = h @ Wg
    gemm_bf16_wmma<0><<<dim3(FF / 128, M / 256), 256, 0, stream>>>(
        hb, Wg_h, M, FF, D, gateF, nullptr, nullptr);

    // 9) act = silu(gate) * (h @ Wu)   [fused epilogue]
    gemm_bf16_wmma<3><<<dim3(FF / 128, M / 256), 256, 0, stream>>>(
        hb, Wu_h, M, FF, D, nullptr, actB, gateF);

    // 10) out = act @ Wd + x1
    gemm_bf16_wmma<2><<<dim3(D / 128, M / 256), 256, 0, stream>>>(
        actB, Wd_h, M, D, FF, out, nullptr, x1);
}
